// Model_22119081574678
// MI455X (gfx1250) — matrix operand order repaired and checked by reading, not compiled, not run on hardware
//
#include <hip/hip_runtime.h>
#include <hip/hip_bf16.h>
#include <math.h>

typedef __attribute__((ext_vector_type(16))) _Float16 v16h;
typedef __attribute__((ext_vector_type(8)))  _Float16 v8h;
typedef __attribute__((ext_vector_type(8)))  float    v8f;
typedef __attribute__((ext_vector_type(4)))  int      v4i;

typedef __attribute__((address_space(1))) v4i ga_v4i;   // global int4
typedef __attribute__((address_space(3))) v4i ls_v4i;   // LDS int4

#define EPS 1e-5f
#define NEGBIG -1000000000.0f

enum Epi { EPI_NONE = 0, EPI_BIAS = 1, EPI_BIAS_RELU = 2, EPI_BIAS_GELU = 3 };

constexpr int LDSW = 48;   // LDS row stride in halfs (96B -> aligned b128 frag reads)

#ifndef __has_builtin
#define __has_builtin(x) 0
#endif
#if __has_builtin(__builtin_amdgcn_global_load_async_to_lds_b128)
#define HAVE_ASYNC 1
#else
#define HAVE_ASYNC 0
#endif

// ---------------------------------------------------------------------------
// Stage one 16B (8-half) segment global->LDS. Async DMA path when available.
// ---------------------------------------------------------------------------
__device__ __forceinline__ void stage16(const _Float16* __restrict__ g, _Float16* l, bool inb) {
    if (inb) {
#if HAVE_ASYNC
        __builtin_amdgcn_global_load_async_to_lds_b128(
            (ga_v4i*)(g), (ls_v4i*)(l), 0, 0);
#else
        *(uint4*)l = *(const uint4*)g;
#endif
    } else {
        uint4 z = {0u, 0u, 0u, 0u};
        *(uint4*)l = z;
    }
}
__device__ __forceinline__ void wait_stage() {
#if HAVE_ASYNC
#if __has_builtin(__builtin_amdgcn_s_wait_asynccnt)
    __builtin_amdgcn_s_wait_asynccnt(0);
#else
    asm volatile("s_wait_asynccnt 0x0" ::: "memory");
#endif
#endif
}

// ---------------------------------------------------------------------------
// WMMA fragment builders (16x16x32 f16, wave32; contiguous k order used for both A and B).
//  A 16x32: lane = row (mod16); k = h + (lane>>4)*16
//           -> one contiguous 16-half run -> two aligned b128 LDS loads
//  B 32x16: lane = col (mod16); k = h + (lane>>4)*16 -> one aligned 32B load
// ---------------------------------------------------------------------------
__device__ __forceinline__ v16h frag_a(const _Float16* As, int row0, int lane) {
    const _Float16* p = As + (row0 + (lane & 15)) * LDSW + ((lane >> 4) << 4);
    v8h lo = *(const v8h*)p;
    v8h hi = *(const v8h*)(p + 8);
    v16h f;
#pragma unroll
    for (int h = 0; h < 8; ++h) { f[h] = lo[h]; f[h + 8] = hi[h]; }
    return f;
}
__device__ __forceinline__ v16h frag_b(const _Float16* Bs, int col0, int lane) {
    const _Float16* p = Bs + (col0 + (lane & 15)) * LDSW + ((lane >> 4) << 4);
    return *(const v16h*)p;
}
__device__ __forceinline__ v8f wmma(v16h a, v16h b, v8f c) {
    return __builtin_amdgcn_wmma_f32_16x16x32_f16(false, a, false, b, (short)0, c, false, false);
}

// ---------------------------------------------------------------------------
// f32 -> f16 cast (weights, once per launch)
// ---------------------------------------------------------------------------
__global__ void cast_f16(const float* __restrict__ s, _Float16* __restrict__ d, int n) {
    int i = blockIdx.x * blockDim.x + threadIdx.x;
    if (i < n) d[i] = (_Float16)s[i];
}

__global__ void bn_fold(const float* __restrict__ b, const float* __restrict__ g,
                        const float* __restrict__ bt, const float* __restrict__ rm,
                        const float* __restrict__ rv, float* __restrict__ alpha,
                        float* __restrict__ beta, int C) {
    int i = blockIdx.x * blockDim.x + threadIdx.x;
    if (i < C) {
        float inv = g[i] * rsqrtf(rv[i] + EPS);
        alpha[i] = inv;
        beta[i]  = (b[i] - rm[i]) * inv + bt[i];
    }
}

// ---------------------------------------------------------------------------
// Implicit-GEMM conv, fused BN+ReLU, f16 NHWC output. 8 waves, each a 32x32
// register tile (4 WMMAs / 32-K step). Block tile (WM*32) x (WN*32).
// ---------------------------------------------------------------------------
template <int WM, int WN, typename AT>
__global__ void conv_wmma(const AT* __restrict__ X, const _Float16* __restrict__ Wt,
                          _Float16* __restrict__ Y, int Nimg, int IC, int IH, int IW,
                          int OC, int KH, int KW, int pad, int nchw,
                          const float* __restrict__ alpha, const float* __restrict__ beta) {
    constexpr int BM = WM * 32, BN = WN * 32;
    const int OH = IH, OW = IW;
    const int M = Nimg * OH * OW;
    const int K = IC * KH * KW;

    __shared__ __align__(32) _Float16 As[BM][LDSW];
    __shared__ __align__(32) _Float16 Bs[BN][LDSW];

    const int tid  = threadIdx.x;
    const int lane = tid & 31;
    const int wave = tid >> 5;
    const int m0 = blockIdx.x * BM;
    const int n0 = blockIdx.y * BN;
    const int wm = (wave / WN) * 32;
    const int wn = (wave % WN) * 32;

    const v8f vz = {0.f, 0.f, 0.f, 0.f, 0.f, 0.f, 0.f, 0.f};
    v8f a00 = vz, a01 = vz, a10 = vz, a11 = vz;

    for (int k0 = 0; k0 < K; k0 += 32) {
        // A: im2col gather (scattered -> elementwise)
        for (int i = tid; i < BM * 32; i += 256) {
            int mm = i >> 5, kk = i & 31;
            int gm = m0 + mm, gk = k0 + kk;
            float val = 0.f;
            if (gm < M && gk < K) {
                int ow = gm % OW; int r = gm / OW;
                int oh = r % OH;  int nimg = r / OH;
                int kw = gk % KW; int r2 = gk / KW;
                int kh = r2 % KH; int ic = r2 / KH;
                int ih = oh + kh - pad, iw = ow + kw - pad;
                if (ih >= 0 && ih < IH && iw >= 0 && iw < IW) {
                    size_t idx = nchw ? ((((size_t)nimg * IC + ic) * IH + ih) * IW + iw)
                                      : ((((size_t)nimg * IH + ih) * IW + iw) * IC + ic);
                    val = (float)X[idx];
                }
            }
            As[mm][kk] = (_Float16)val;
        }
        // B: f16 weights [OC,K] row-major (elementwise; K not 8-aligned for conv1)
        for (int i = tid; i < BN * 32; i += 256) {
            int nn = i >> 5, kk = i & 31;
            int gn = n0 + nn, gk = k0 + kk;
            Bs[nn][kk] = (gn < OC && gk < K) ? Wt[(size_t)gn * K + gk] : (_Float16)0.f;
        }
        {   // prefetch next weight K-tile (linear, speculative)
            int nk = k0 + 32 + (tid & 31);
            int gn = n0 + (tid >> 5);
            if (gn < OC && nk < K) __builtin_prefetch(&Wt[(size_t)gn * K + nk], 0, 1);
        }
        __syncthreads();

        v16h fa0 = frag_a(&As[0][0], wm,      lane);
        v16h fa1 = frag_a(&As[0][0], wm + 16, lane);
        v16h fb0 = frag_b(&Bs[0][0], wn,      lane);
        v16h fb1 = frag_b(&Bs[0][0], wn + 16, lane);
        a00 = wmma(fa0, fb0, a00);
        a01 = wmma(fa0, fb1, a01);
        a10 = wmma(fa1, fb0, a10);
        a11 = wmma(fa1, fb1, a11);
        __syncthreads();
    }

    const int ml = (lane >> 4) << 3;
    const int nl = lane & 15;
    v8f* accs[2][2] = {{&a00, &a01}, {&a10, &a11}};
#pragma unroll
    for (int i = 0; i < 2; ++i) {
#pragma unroll
        for (int j = 0; j < 2; ++j) {
            int oc = n0 + wn + j * 16 + nl;
            int mb = m0 + wm + i * 16 + ml;
            if (oc < OC) {
                float a = alpha[oc], bb = beta[oc];
                v8f& ac = *accs[i][j];
#pragma unroll
                for (int v = 0; v < 8; ++v) {
                    int m = mb + v;
                    if (m < M) Y[(size_t)m * OC + oc] = (_Float16)fmaxf(ac[v] * a + bb, 0.f);
                }
            }
        }
    }
}

// ---------------------------------------------------------------------------
// Generic GEMM: C[M,N] = A[M,K] * W[N,K]^T (+ epilogue). A,W are f16 with
// K % 32 == 0 -> staging is pure 16B segments (async DMA when available).
// ---------------------------------------------------------------------------
template <int WM, int WN, typename OT>
__global__ void gemm_nt_wmma(const _Float16* __restrict__ A, const _Float16* __restrict__ Wt,
                             OT* __restrict__ C, int M, int N, int K,
                             const float* __restrict__ bias, int epi) {
    constexpr int BM = WM * 32, BN = WN * 32;
    __shared__ __align__(32) _Float16 As[BM][LDSW];
    __shared__ __align__(32) _Float16 Bs[BN][LDSW];

    const int tid  = threadIdx.x;
    const int lane = tid & 31;
    const int wave = tid >> 5;
    const int m0 = blockIdx.x * BM;
    const int n0 = blockIdx.y * BN;
    const int wm = (wave / WN) * 32;
    const int wn = (wave % WN) * 32;

    const v8f vz = {0.f, 0.f, 0.f, 0.f, 0.f, 0.f, 0.f, 0.f};
    v8f a00 = vz, a01 = vz, a10 = vz, a11 = vz;

    for (int k0 = 0; k0 < K; k0 += 32) {
        for (int i = tid; i < BM * 4; i += 256) {
            int mm = i >> 2, seg = (i & 3) << 3;
            stage16(&A[(size_t)(m0 + mm) * K + k0 + seg], &As[mm][seg], (m0 + mm) < M);
        }
        for (int i = tid; i < BN * 4; i += 256) {
            int nn = i >> 2, seg = (i & 3) << 3;
            stage16(&Wt[(size_t)(n0 + nn) * K + k0 + seg], &Bs[nn][seg], (n0 + nn) < N);
        }
        wait_stage();
        __syncthreads();

        v16h fa0 = frag_a(&As[0][0], wm,      lane);
        v16h fa1 = frag_a(&As[0][0], wm + 16, lane);
        v16h fb0 = frag_b(&Bs[0][0], wn,      lane);
        v16h fb1 = frag_b(&Bs[0][0], wn + 16, lane);
        a00 = wmma(fa0, fb0, a00);
        a01 = wmma(fa0, fb1, a01);
        a10 = wmma(fa1, fb0, a10);
        a11 = wmma(fa1, fb1, a11);
        __syncthreads();
    }

    const int ml = (lane >> 4) << 3;
    const int nl = lane & 15;
    v8f* accs[2][2] = {{&a00, &a01}, {&a10, &a11}};
#pragma unroll
    for (int i = 0; i < 2; ++i) {
#pragma unroll
        for (int j = 0; j < 2; ++j) {
            int n = n0 + wn + j * 16 + nl;
            int mb = m0 + wm + i * 16 + ml;
            if (n < N) {
                float b = (epi != EPI_NONE && bias) ? bias[n] : 0.f;
                v8f& ac = *accs[i][j];
#pragma unroll
                for (int v = 0; v < 8; ++v) {
                    int m = mb + v;
                    if (m < M) {
                        float x = ac[v] + b;
                        if (epi == EPI_BIAS_RELU)      x = fmaxf(x, 0.f);
                        else if (epi == EPI_BIAS_GELU) x = 0.5f * x * (1.f + erff(x * 0.70710678118f));
                        C[(size_t)m * N + n] = (OT)x;
                    }
                }
            }
        }
    }
}

// ---------------------------------------------------------------------------
// Average pool (NHWC, f16 in/out). chwflat=1 -> CHW-flattened features.
// ---------------------------------------------------------------------------
template <typename T>
__global__ void avgpool_nhwc(const T* __restrict__ X, T* __restrict__ Y,
                             int Nimg, int H, int W, int C, int k, int chwflat) {
    const int OH = H / k, OW = W / k;
    long long total = (long long)Nimg * OH * OW * C;
    long long idx = (long long)blockIdx.x * blockDim.x + threadIdx.x;
    if (idx >= total) return;
    int c = (int)(idx % C); long long r = idx / C;
    int ow = (int)(r % OW); r /= OW;
    int oh = (int)(r % OH); int n = (int)(r / OH);
    float s = 0.f;
    for (int dy = 0; dy < k; ++dy)
        for (int dx = 0; dx < k; ++dx)
            s += (float)X[(((size_t)n * H + (oh * k + dy)) * W + (ow * k + dx)) * C + c];
    s *= 1.f / (float)(k * k);
    if (chwflat)
        Y[(size_t)n * C * OH * OW + (size_t)c * OH * OW + (size_t)oh * OW + ow] = (T)s;
    else
        Y[(((size_t)n * OH + oh) * OW + ow) * C + c] = (T)s;
}

// ---------------------------------------------------------------------------
// GRU direction scan. One block per batch, 3H (=384) threads. f32 throughout.
// ---------------------------------------------------------------------------
__global__ void gru_scan(const float* __restrict__ xg, const float* __restrict__ whh,
                         const float* __restrict__ bhh, float* __restrict__ hcat,
                         int S, int H, int reverse, int outoff) {
    const int b = blockIdx.x;
    const int j = threadIdx.x;
    __shared__ float h[128];
    __shared__ float hg[384];
    if (j < H) h[j] = 0.f;
    __syncthreads();
    for (int t = 0; t < S; ++t) {
        int s = reverse ? (S - 1 - t) : t;
        float acc = bhh[j];
        const float* wr = &whh[(size_t)j * H];
        for (int d = 0; d < H; ++d) acc += wr[d] * h[d];
        hg[j] = acc;
        __syncthreads();
        if (j < H) {
            const float* xr = &xg[((size_t)(b * S + s)) * (3 * H)];
            float r  = 1.f / (1.f + __expf(-(xr[j]         + hg[j])));
            float z  = 1.f / (1.f + __expf(-(xr[H + j]     + hg[H + j])));
            float nn = tanhf(xr[2 * H + j] + r * hg[2 * H + j]);
            float hn = (1.f - z) * nn + z * h[j];
            h[j] = hn;
            hcat[((size_t)(b * S + s)) * (2 * H) + outoff + j] = hn;
        }
        __syncthreads();
    }
}

// ---------------------------------------------------------------------------
// LayerNorm (f32 in, f16 out; block = D threads, one block per row)
// ---------------------------------------------------------------------------
__global__ void layernorm_k(const float* __restrict__ x, const float* __restrict__ g,
                            const float* __restrict__ b, _Float16* __restrict__ y, int D) {
    const int row = blockIdx.x;
    const int t = threadIdx.x;
    __shared__ float red[256];
    float v = x[(size_t)row * D + t];
    red[t] = v; __syncthreads();
    for (int s = D >> 1; s > 0; s >>= 1) { if (t < s) red[t] += red[t + s]; __syncthreads(); }
    float mean = red[0] / (float)D;
    __syncthreads();
    float d = v - mean;
    red[t] = d * d; __syncthreads();
    for (int s = D >> 1; s > 0; s >>= 1) { if (t < s) red[t] += red[t + s]; __syncthreads(); }
    float var = red[0] / (float)D;
    y[(size_t)row * D + t] = (_Float16)(d * rsqrtf(var + EPS) * g[t] + b[t]);
}

// ---------------------------------------------------------------------------
// Attention with ALiBi (f16 q/k/v -> f16 out). grid (S, nh, B), block S=128.
// ---------------------------------------------------------------------------
__global__ void attention_k(const _Float16* __restrict__ q, const _Float16* __restrict__ k,
                            const _Float16* __restrict__ v, _Float16* __restrict__ out,
                            int S, int nh, int hd) {
    const int b = blockIdx.z, head = blockIdx.y, qi = blockIdx.x;
    const int t = threadIdx.x;
    const int D = nh * hd;
    __shared__ float qv[64];
    __shared__ float sc[128];
    __shared__ float red[128];
    if (t < hd) qv[t] = (float)q[((size_t)(b * S + qi)) * D + head * hd + t];
    __syncthreads();

    const _Float16* kr = &k[((size_t)(b * S + t)) * D + head * hd];
    float s = 0.f;
    for (int d = 0; d < hd; ++d) s += qv[d] * (float)kr[d];
    s *= rsqrtf((float)hd);

    const int half = nh >> 1;
    const float base = powf(24.f, 1.f / (float)half);
    float rel = (float)(t - qi);
    float bias;
    if (head < half) {
        float slope = powf(base, -(float)(head + 1));
        bias = (t <= qi) ? slope * rel : NEGBIG;
    } else {
        float slope = powf(base, -(float)(head - half + 1));
        bias = (t >= qi) ? -slope * rel : NEGBIG;
    }
    s += bias;

    sc[t] = s; red[t] = s; __syncthreads();
    for (int st = S >> 1; st > 0; st >>= 1) { if (t < st) red[t] = fmaxf(red[t], red[t + st]); __syncthreads(); }
    float mx = red[0]; __syncthreads();
    float p = __expf(s - mx);
    sc[t] = p; red[t] = p; __syncthreads();
    for (int st = S >> 1; st > 0; st >>= 1) { if (t < st) red[t] += red[t + st]; __syncthreads(); }
    float inv = 1.f / red[0];
    __syncthreads();

    if (t < hd) {
        float o = 0.f;
        for (int kk = 0; kk < S; ++kk)
            o += sc[kk] * (float)v[((size_t)(b * S + kk)) * D + head * hd + t];
        out[((size_t)(b * S + qi)) * D + head * hd + t] = (_Float16)(o * inv);
    }
}

__global__ void add_k(float* __restrict__ y, const float* __restrict__ x, int n) {
    int i = blockIdx.x * blockDim.x + threadIdx.x;
    if (i < n) y[i] += x[i];
}

// ---------------------------------------------------------------------------
// Host orchestration
// ---------------------------------------------------------------------------
extern "C" void kernel_launch(void* const* d_in, const int* in_sizes, int n_in,
                              void* d_out, int out_size, void* d_ws, size_t ws_size,
                              hipStream_t stream) {
    (void)in_sizes; (void)out_size; (void)ws_size;
    const float* F[200];
    for (int i = 0; i < n_in && i < 200; ++i) F[i] = (const float*)d_in[i];

    const float* x = F[0];
    const int CONV = 1;
    const int GRU  = CONV + 18;
    const int TFM  = GRU + 8;
    const int FC   = TFM + 98;

    const int B = 2, S = 128, D = 256, nh = 4, hd = 64;
    const int NI = B * S;

    // byte-based workspace carver, 64B aligned
    char* base = (char*)d_ws;
    size_t off = 0;
    auto carveB = [&](size_t bytes) { off = (off + 63) & ~(size_t)63; void* p = base + off; off += bytes; return p; };
    auto carveH = [&](size_t n) { return (_Float16*)carveB(n * sizeof(_Float16)); };
    auto carveF = [&](size_t n) { return (float*)carveB(n * sizeof(float)); };

    // f16 activations
    _Float16* c1  = carveH((size_t)NI * 108 * 108 * 32);
    _Float16* p1  = carveH((size_t)NI * 36 * 36 * 32);
    _Float16* c2  = carveH((size_t)NI * 36 * 36 * 128);
    _Float16* p2  = carveH((size_t)NI * 12 * 12 * 128);
    _Float16* c3  = carveH((size_t)NI * 12 * 12 * 512);
    _Float16* yf  = carveH((size_t)NI * 8192);
    _Float16* hn  = carveH((size_t)NI * D);
    _Float16* qb  = carveH((size_t)NI * D);
    _Float16* kb  = carveH((size_t)NI * D);
    _Float16* vb  = carveH((size_t)NI * D);
    _Float16* at  = carveH((size_t)NI * D);
    _Float16* ff  = carveH((size_t)NI * 1024);
    _Float16* f1  = carveH((size_t)NI * 64);
    // f32 buffers
    float* xgf = carveF((size_t)NI * 384);
    float* xgb = carveF((size_t)NI * 384);
    float* h   = carveF((size_t)NI * D);
    float* tb  = carveF((size_t)NI * D);
    float* al  = carveF(512);
    float* be  = carveF(512);

    auto cdiv = [](long long a, long long b) { return (unsigned)((a + b - 1) / b); };

    // ---- cast all matmul weights to f16 once ----
    auto castW = [&](const float* src, size_t n) {
        _Float16* d = carveH(n);
        cast_f16<<<cdiv((long long)n, 256), 256, 0, stream>>>(src, d, (int)n);
        return d;
    };
    _Float16* w1h = castW(F[CONV+0],  (size_t)32 * 2 * 49);
    _Float16* w2h = castW(F[CONV+6],  (size_t)128 * 32 * 25);
    _Float16* w3h = castW(F[CONV+12], (size_t)512 * 128 * 25);
    _Float16* wihf = castW(F[GRU+0], (size_t)384 * 8192);
    _Float16* wihb = castW(F[GRU+4], (size_t)384 * 8192);
    _Float16* tw[6][6];
    for (int l = 0; l < 6; ++l) {
        const int L = TFM + l * 16;
        tw[l][0] = castW(F[L+0],  (size_t)D * D);        // wq
        tw[l][1] = castW(F[L+2],  (size_t)D * D);        // wk
        tw[l][2] = castW(F[L+4],  (size_t)D * D);        // wv
        tw[l][3] = castW(F[L+6],  (size_t)D * D);        // wo
        tw[l][4] = castW(F[L+8],  (size_t)1024 * D);     // w_ff1
        tw[l][5] = castW(F[L+10], (size_t)D * 1024);     // w_ff2
    }
    _Float16* fw1 = castW(F[FC+0], (size_t)64 * D);
    _Float16* fw2 = castW(F[FC+2], (size_t)2 * 64);

    // ---- conv1: 2->32, 7x7 pad3 @108, NCHW f32 input; tile 256x32 ----
    bn_fold<<<1, 512, 0, stream>>>(F[CONV+1], F[CONV+2], F[CONV+3], F[CONV+4], F[CONV+5], al, be, 32);
    {
        long long M = (long long)NI * 108 * 108;
        conv_wmma<8,1,float><<<dim3(cdiv(M, 256), 1), 256, 0, stream>>>(
            x, w1h, c1, NI, 2, 108, 108, 32, 7, 7, 3, 1, al, be);
    }
    avgpool_nhwc<_Float16><<<cdiv((long long)NI*36*36*32, 256), 256, 0, stream>>>(c1, p1, NI, 108, 108, 32, 3, 0);

    // ---- conv2: 32->128, 5x5 pad2 @36; tile 64x128 ----
    bn_fold<<<1, 512, 0, stream>>>(F[CONV+7], F[CONV+8], F[CONV+9], F[CONV+10], F[CONV+11], al, be, 128);
    {
        long long M = (long long)NI * 36 * 36;
        conv_wmma<2,4,_Float16><<<dim3(cdiv(M, 64), 1), 256, 0, stream>>>(
            p1, w2h, c2, NI, 32, 36, 36, 128, 5, 5, 2, 0, al, be);
    }
    avgpool_nhwc<_Float16><<<cdiv((long long)NI*12*12*128, 256), 256, 0, stream>>>(c2, p2, NI, 36, 36, 128, 3, 0);

    // ---- conv3: 128->512, 5x5 pad2 @12 ----
    bn_fold<<<1, 512, 0, stream>>>(F[CONV+13], F[CONV+14], F[CONV+15], F[CONV+16], F[CONV+17], al, be, 512);
    {
        long long M = (long long)NI * 12 * 12;
        conv_wmma<2,4,_Float16><<<dim3(cdiv(M, 64), 4), 256, 0, stream>>>(
            p2, w3h, c3, NI, 128, 12, 12, 512, 5, 5, 2, 0, al, be);
    }
    avgpool_nhwc<_Float16><<<cdiv((long long)NI*4*4*512, 256), 256, 0, stream>>>(c3, yf, NI, 12, 12, 512, 3, 1);

    // ---- GRU input projections (f16 A/W, f32 out) ----
    {
        dim3 g(cdiv(NI, 64), cdiv(384, 128));
        gemm_nt_wmma<2,4,float><<<g, 256, 0, stream>>>(yf, wihf, xgf, NI, 384, 8192, F[GRU+2], EPI_BIAS);
        gemm_nt_wmma<2,4,float><<<g, 256, 0, stream>>>(yf, wihb, xgb, NI, 384, 8192, F[GRU+6], EPI_BIAS);
    }
    gru_scan<<<B, 384, 0, stream>>>(xgf, F[GRU+1], F[GRU+3], h, S, 128, 0, 0);
    gru_scan<<<B, 384, 0, stream>>>(xgb, F[GRU+5], F[GRU+7], h, S, 128, 1, 128);

    // ---- Transformer ----
    dim3 gDD(cdiv(NI, 64), cdiv(D, 128));
    dim3 gFD(cdiv(NI, 64), cdiv(1024, 128));
    for (int l = 0; l < 6; ++l) {
        const int L = TFM + l * 16;
        layernorm_k<<<NI, D, 0, stream>>>(h, F[L+12], F[L+13], hn, D);
        gemm_nt_wmma<2,4,_Float16><<<gDD, 256, 0, stream>>>(hn, tw[l][0], qb, NI, D, D, F[L+1], EPI_BIAS);
        gemm_nt_wmma<2,4,_Float16><<<gDD, 256, 0, stream>>>(hn, tw[l][1], kb, NI, D, D, F[L+3], EPI_BIAS);
        gemm_nt_wmma<2,4,_Float16><<<gDD, 256, 0, stream>>>(hn, tw[l][2], vb, NI, D, D, F[L+5], EPI_BIAS);
        attention_k<<<dim3(S, nh, B), S, 0, stream>>>(qb, kb, vb, at, S, nh, hd);
        gemm_nt_wmma<2,4,float><<<gDD, 256, 0, stream>>>(at, tw[l][3], tb, NI, D, D, F[L+7], EPI_BIAS);
        add_k<<<cdiv(NI*D, 256), 256, 0, stream>>>(h, tb, NI * D);

        layernorm_k<<<NI, D, 0, stream>>>(h, F[L+14], F[L+15], hn, D);
        gemm_nt_wmma<2,4,_Float16><<<gFD, 256, 0, stream>>>(hn, tw[l][4], ff, NI, 1024, D, F[L+9], EPI_BIAS_GELU);
        gemm_nt_wmma<2,4,float><<<gDD, 256, 0, stream>>>(ff, tw[l][5], tb, NI, D, 1024, F[L+11], EPI_BIAS);
        add_k<<<cdiv(NI*D, 256), 256, 0, stream>>>(h, tb, NI * D);
    }

    // ---- final LN + FC head ----
    layernorm_k<<<NI, D, 0, stream>>>(h, F[TFM+96], F[TFM+97], hn, D);
    {
        gemm_nt_wmma<8,1,_Float16><<<dim3(cdiv(NI, 256), 2), 256, 0, stream>>>(
            hn, fw1, f1, NI, 64, D, F[FC+1], EPI_BIAS_RELU);
        gemm_nt_wmma<8,1,float><<<dim3(cdiv(NI, 256), 1), 256, 0, stream>>>(
            f1, fw2, (float*)d_out, NI, 2, 64, F[FC+3], EPI_BIAS);
    }
}
